// FullAttention_42975442764617
// MI455X (gfx1250) — matrix operand order repaired and checked by reading, not compiled, not run on hardware
//
#include <hip/hip_runtime.h>

typedef __attribute__((ext_vector_type(16))) _Float16 v16h;
typedef __attribute__((ext_vector_type(8)))  float    v8f;

constexpr int Bb = 2, Hh = 16, Ss = 2048, Cc = 64;
constexpr int MT = 128;          // query rows per workgroup (8 waves x 16 rows)
constexpr int NT = 64;           // keys per LDS-staged block
constexpr int NWAVE = MT / 16;   // 8 waves, 256 threads, wave32
constexpr int PAD = 8;           // f16 row pad to spread LDS banks

__global__ __launch_bounds__(256) void fa_causal_wmma(
    const float* __restrict__ Qp, const float* __restrict__ Kp,
    const float* __restrict__ Vp, float* __restrict__ Op)
{
  __shared__ _Float16 ldsK[NT][Cc + PAD];          // [key][chan], row-major
  __shared__ _Float16 ldsV[Cc][NT + PAD];          // transposed: [chan][key]
  __shared__ _Float16 ldsP[NWAVE][16][NT + PAD];   // per-wave P relayout scratch

  const int bh   = blockIdx.y;
  const int qb   = blockIdx.x * MT;
  const int tid  = threadIdx.x;
  const int lane = tid & 31;                       // wave32
  const int wave = tid >> 5;
  const int nloc = lane & 15;
  const int hi   = lane >> 4;

  const size_t base  = (size_t)bh * Ss * Cc;
  const int    qrow0 = qb + wave * 16;

  // ---- Q A-fragments (16x32 f16 layout), pre-scaled by 1/sqrt(C)=0.125 ----
  v16h qa[2];
  {
    const float* Qrow = Qp + base + (size_t)(qrow0 + nloc) * Cc;
    #pragma unroll
    for (int kc = 0; kc < 2; ++kc)
      #pragma unroll
      for (int v = 0; v < 8; ++v) {
        const int kk = kc * 32 + ((v >> 2) << 4) + ((v & 3) << 1) + (hi << 3);
        qa[kc][2 * v]     = (_Float16)(Qrow[kk]     * 0.125f);
        qa[kc][2 * v + 1] = (_Float16)(Qrow[kk + 1] * 0.125f);
      }
  }

  float mrow[8], lrow[8];
  v8f   o[4];
  #pragma unroll
  for (int r = 0; r < 8; ++r) { mrow[r] = -1e30f; lrow[r] = 0.f; }
  #pragma unroll
  for (int j = 0; j < 4; ++j) o[j] = {};

  const int kb_end = (qb + MT) / NT;   // causal: keys up to last row of block

  for (int kb = 0; kb < kb_end; ++kb) {
    const float* Kg = Kp + base + (size_t)kb * NT * Cc;
    const float* Vg = Vp + base + (size_t)kb * NT * Cc;
    if (kb + 1 < kb_end) {             // global_prefetch_b8 for next tiles
      __builtin_prefetch(Kg + NT * Cc, 0, 1);
      __builtin_prefetch(Vg + NT * Cc, 0, 1);
    }

    __syncthreads();
    {   // cooperative stage: 64x64 fp32 -> f16; K row-major, V transposed
      const float4* Kg4 = (const float4*)Kg;
      const float4* Vg4 = (const float4*)Vg;
      #pragma unroll
      for (int i = 0; i < 4; ++i) {
        const int idx4 = tid + i * 256;          // 1024 float4 = 4096 elems
        const int n = idx4 >> 4;
        const int c = (idx4 & 15) << 2;
        const float4 kv = Kg4[idx4];
        ldsK[n][c + 0] = (_Float16)kv.x; ldsK[n][c + 1] = (_Float16)kv.y;
        ldsK[n][c + 2] = (_Float16)kv.z; ldsK[n][c + 3] = (_Float16)kv.w;
        const float4 vv = Vg4[idx4];
        ldsV[c + 0][n] = (_Float16)vv.x; ldsV[c + 1][n] = (_Float16)vv.y;
        ldsV[c + 2][n] = (_Float16)vv.z; ldsV[c + 3][n] = (_Float16)vv.w;
      }
    }
    __syncthreads();

    // ---- S = Q K^T : 4 key tiles, C=64 reduced as 2 WMMAs of K=32 ----
    v8f s4[4];
    #pragma unroll
    for (int t = 0; t < 4; ++t) {
      v8f acc = {};
      #pragma unroll
      for (int kc = 0; kc < 2; ++kc) {
        v16h bf;   // B 32x16 layout: K = 16*(v/4) + 8*hi + 2*(v%4) (+ kc*32), N = nloc
        #pragma unroll
        for (int v = 0; v < 8; ++v) {
          const int kk = kc * 32 + ((v >> 2) << 4) + ((v & 3) << 1) + (hi << 3);
          bf[2 * v]     = ldsK[t * 16 + nloc][kk];
          bf[2 * v + 1] = ldsK[t * 16 + nloc][kk + 1];
        }
        acc = __builtin_amdgcn_wmma_f32_16x16x32_f16(false, qa[kc], false, bf,
                                                     (short)0, acc, false, false);
      }
      s4[t] = acc;
    }

    // ---- causal mask + online softmax (16-lane row groups) ----
    const bool need_mask = (kb * NT + NT - 1) > qrow0;
    float alpha[8];
    #pragma unroll
    for (int r = 0; r < 8; ++r) {
      const int qrow = qrow0 + (hi << 3) + r;
      float mx = -1e30f;
      #pragma unroll
      for (int t = 0; t < 4; ++t) {
        float val = s4[t][r];
        if (need_mask && (kb * NT + t * 16 + nloc) > qrow) { val = -1e30f; s4[t][r] = val; }
        mx = fmaxf(mx, val);
      }
      #pragma unroll
      for (int off = 1; off < 16; off <<= 1)
        mx = fmaxf(mx, __shfl_xor(mx, off, 32));
      const float mnew = fmaxf(mrow[r], mx);
      alpha[r] = __expf(mrow[r] - mnew);
      mrow[r]  = mnew;
      float rs = 0.f;
      #pragma unroll
      for (int t = 0; t < 4; ++t) {
        const float p = __expf(s4[t][r] - mnew);
        s4[t][r] = p;
        rs += p;
      }
      #pragma unroll
      for (int off = 1; off < 16; off <<= 1)
        rs += __shfl_xor(rs, off, 32);
      lrow[r] = lrow[r] * alpha[r] + rs;
    }
    #pragma unroll
    for (int j = 0; j < 4; ++j)
      #pragma unroll
      for (int r = 0; r < 8; ++r)
        o[j][r] *= alpha[r];

    // ---- P: C-layout f32 -> (wave-private LDS) -> A-layout f16 ----
    #pragma unroll
    for (int t = 0; t < 4; ++t)
      #pragma unroll
      for (int r = 0; r < 8; ++r)
        ldsP[wave][(hi << 3) + r][t * 16 + nloc] = (_Float16)s4[t][r];
    // same-wave LDS ops are in-order (DScnt) -> no barrier needed
    v16h pa[2];
    #pragma unroll
    for (int kc = 0; kc < 2; ++kc)
      #pragma unroll
      for (int v = 0; v < 8; ++v) {
        const int kk = kc * 32 + ((v >> 2) << 4) + ((v & 3) << 1) + (hi << 3);
        pa[kc][2 * v]     = ldsP[wave][nloc][kk];
        pa[kc][2 * v + 1] = ldsP[wave][nloc][kk + 1];
      }

    // ---- O += P V : 4 output-channel tiles, 64 keys as 2 WMMAs of K=32 ----
    #pragma unroll
    for (int j = 0; j < 4; ++j) {
      #pragma unroll
      for (int kc = 0; kc < 2; ++kc) {
        v16h vb;   // B layout from transposed V: contiguous key pairs
        #pragma unroll
        for (int v = 0; v < 8; ++v) {
          const int kk = kc * 32 + ((v >> 2) << 4) + ((v & 3) << 1) + (hi << 3);
          vb[2 * v]     = ldsV[j * 16 + nloc][kk];
          vb[2 * v + 1] = ldsV[j * 16 + nloc][kk + 1];
        }
        o[j] = __builtin_amdgcn_wmma_f32_16x16x32_f16(false, pa[kc], false, vb,
                                                      (short)0, o[j], false, false);
      }
    }
  }

  // ---- epilogue: normalize and store fp32 ----
  #pragma unroll
  for (int r = 0; r < 8; ++r) {
    const float  inv = 1.f / lrow[r];
    const size_t row = base + (size_t)(qrow0 + (hi << 3) + r) * Cc;
    #pragma unroll
    for (int j = 0; j < 4; ++j)
      Op[row + j * 16 + nloc] = o[j][r] * inv;
  }
}

extern "C" void kernel_launch(void* const* d_in, const int* in_sizes, int n_in,
                              void* d_out, int out_size, void* d_ws, size_t ws_size,
                              hipStream_t stream) {
  (void)in_sizes; (void)n_in; (void)d_ws; (void)ws_size; (void)out_size;
  const float* q = (const float*)d_in[0];
  const float* k = (const float*)d_in[1];
  const float* v = (const float*)d_in[2];
  float* out = (float*)d_out;
  dim3 grid(Ss / MT, Bb * Hh);   // (16, 32)
  dim3 block(256);               // 8 wave32 waves
  fa_causal_wmma<<<grid, block, 0, stream>>>(q, k, v, out);
}
